// MultiheadAttention_8280696947425
// MI455X (gfx1250) — matrix operand order repaired and checked by reading, not compiled, not run on hardware
//
#include <hip/hip_runtime.h>

// ---------------------------------------------------------------------------
// Multi-head self-attention forward for MI455X (gfx1250, wave32, WMMA + TDM).
// T=2048, B=2, E=1024, H=16, D=64.  f16 WMMA (16x16x32) with f32 accumulate.
// Projection GEMMs: TDM (tensor_load_to_lds) stages 64x32 weight panels into
// double-buffered LDS, overlapped with WMMA compute; s_wait_tensorcnt syncs.
// ---------------------------------------------------------------------------

typedef __attribute__((ext_vector_type(16))) _Float16 v16h;
typedef __attribute__((ext_vector_type(8)))  _Float16 v8h;
typedef __attribute__((ext_vector_type(8)))  float    v8f;
typedef __attribute__((ext_vector_type(4)))  unsigned int v4u;
typedef __attribute__((ext_vector_type(8)))  int      v8i;
typedef __attribute__((ext_vector_type(4)))  int      v4i;

#define T_SEQ 2048
#define BATCH 2
#define EMB   1024
#define HEADS 16
#define DHEAD 64
#define NROWS (T_SEQ * BATCH)   // 4096 token rows

// ---------------------------------------------------------------------------
// WMMA helper:  D(16x16 f32) = A(16x32 f16) * B(32x16 f16) + C
// ---------------------------------------------------------------------------
__device__ __forceinline__ v8f wmma16(v16h a, v16h b, v8f c) {
  return __builtin_amdgcn_wmma_f32_16x16x32_f16(
      false, a, false, b, (short)0, c, false, false);
}

// ---------------------------------------------------------------------------
// A-fragment (16x32, f16, row-major source; src points at element (m0,k0)).
// lanes 0-15: row M=lane, K={0..15}; lanes 16-31: row M=lane-16,
// K={16..31}.  Two 16-byte loads per lane.
// ---------------------------------------------------------------------------
__device__ __forceinline__ v16h load_frag_a(const _Float16* __restrict__ src, int lda) {
  const int lane = threadIdx.x & 31;
  const _Float16* row = src + (size_t)(lane & 15) * lda + ((lane >> 4) << 4);
  v8h lo = *(const v8h*)(row);
  v8h hi = *(const v8h*)(row + 8);
  return __builtin_shufflevector(lo, hi, 0, 1, 2, 3, 4, 5, 6, 7,
                                         8, 9, 10, 11, 12, 13, 14, 15);
}

// ---------------------------------------------------------------------------
// B-fragment (32x16, f16). Source is "N-major, K-contiguous": element (k,n)
// at src[n*ldb + k] (src points at element (k0,n0)).  One 32B load per lane.
// ---------------------------------------------------------------------------
__device__ __forceinline__ v16h load_frag_b(const _Float16* __restrict__ src, int ldb) {
  const int lane = threadIdx.x & 31;
  return *(const v16h*)(src + (size_t)(lane & 15) * ldb + ((lane >> 4) << 4));
}

// ---------------------------------------------------------------------------
// TDM: DMA a 64-row x 32-col f16 tile (row stride 1024 elements) from global
// into LDS at byte offset lds_off.  D# per CDNA5 ISA.
//   group0: count=1 | lds_addr | global_addr | type=2
//   group1: data_size=2B, tensor_dim0/1=1024, tile_dim0=32, tile_dim1=64,
//           tensor_dim0_stride=1024
// Tracked with TENSORcnt; wave-level op (EXEC ignored).
// ---------------------------------------------------------------------------
__device__ __forceinline__ void tdm_load_64x32(unsigned lds_off,
                                               const _Float16* gsrc) {
  unsigned long long ga = (unsigned long long)gsrc;
  v4u g0;
  g0[0] = 1u;                                        // count=1, user mode
  g0[1] = lds_off;                                   // LDS byte address
  g0[2] = (unsigned)(ga & 0xFFFFFFFFu);              // global_addr[31:0]
  g0[3] = (unsigned)((ga >> 32) & 0x1FFFFFFu)        // global_addr[56:32]
        | (2u << 30);                                // type=2 ("image")
  v8i g1;
  g1[0] = (int)0x00010000;          // wg_mask=0, data_size=1 (2 bytes)
  g1[1] = (int)(1024u << 16);       // tensor_dim0[15:0] -> bits[63:48]
  g1[2] = (int)(1024u << 16);       // tensor_dim0 hi=0 | tensor_dim1 lo
  g1[3] = (int)(32u << 16);         // tensor_dim1 hi=0 | tile_dim0=32
  g1[4] = 64;                       // tile_dim1=64, tile_dim2=0
  g1[5] = 1024;                     // tensor_dim0_stride[31:0]
  g1[6] = 0;                        // stride0 hi | stride1 lo
  g1[7] = 0;                        // stride1 hi
  v4i z4 = {0, 0, 0, 0};
#if defined(__clang_major__) && (__clang_major__ >= 23)
  v8i z8 = {0, 0, 0, 0, 0, 0, 0, 0};
  __builtin_amdgcn_tensor_load_to_lds(g0, g1, z4, z4, z8, 0);
#else
  __builtin_amdgcn_tensor_load_to_lds(g0, g1, z4, z4, 0);
#endif
}

// ---------------------------------------------------------------------------
// f32 -> f16 bulk convert (grid-stride)
// ---------------------------------------------------------------------------
__global__ void f32_to_f16_kernel(const float* __restrict__ src,
                                  _Float16* __restrict__ dst, int n) {
  for (int i = blockIdx.x * blockDim.x + threadIdx.x; i < n;
       i += gridDim.x * blockDim.x)
    dst[i] = (_Float16)src[i];
}

// ---------------------------------------------------------------------------
// Fused QKV projection.  Y = X @ W^T + b.
// Block = 128 threads (4 waves) -> 64x64 output tile; blockIdx.y picks Q/K/V.
// Weight panels (64 e-rows x 32 k) staged by TDM into double-buffered LDS,
// shared by all 4 waves; A fragments register-double-buffered from global.
//   Q -> qh[(b*H+h)*T*64 + t*64 + d]  (pre-scaled by D^-0.5)
//   K -> kh[(b*H+h)*T*64 + t*64 + d]
//   V -> vt[(b*H+h)*64*T + d*T + t]   (transposed for PV B-fragments)
// ---------------------------------------------------------------------------
__global__ void qkv_gemm_kernel(const _Float16* __restrict__ Xh,
                                const _Float16* __restrict__ Wqh,
                                const _Float16* __restrict__ Wkh,
                                const _Float16* __restrict__ Wvh,
                                const float* __restrict__ bq,
                                const float* __restrict__ bk,
                                const float* __restrict__ bv,
                                _Float16* __restrict__ qh,
                                _Float16* __restrict__ kh,
                                _Float16* __restrict__ vt) {
  __shared__ __align__(32) _Float16 bt[2][64 * 32];   // 2 x 4KB weight panels

  const int which = blockIdx.y;                      // 0=Q 1=K 2=V
  const _Float16* W = (which == 0) ? Wqh : (which == 1) ? Wkh : Wvh;
  const float* bias = (which == 0) ? bq : (which == 1) ? bk : bv;

  const int wave = threadIdx.x >> 5;
  const int row0 = (blockIdx.x >> 4) * 64;           // 64 row-tiles
  const int e0 = (blockIdx.x & 15) * 64;             // 16 col-groups
  const int n0 = row0 + wave * 16;

  const int lane = threadIdx.x & 31;
  const int l = lane & 15, hiH = lane >> 4;
  const unsigned lds_base = (unsigned)(unsigned long long)&bt[0][0];

  const _Float16* arow = Xh + (size_t)n0 * EMB;
  const _Float16* wbase = W + (size_t)e0 * EMB;

  v8f c0 = {}, c1 = {}, c2 = {}, c3 = {};

  // prologue: kick off panel 0, prefetch first A fragment
  if (wave == 0) tdm_load_64x32(lds_base, wbase);
  v16h a_next = load_frag_a(arow, EMB);

  const int NC = EMB / 32;                           // 32 K-chunks
  for (int kc = 0; kc < NC; ++kc) {
    if (wave == 0) __builtin_amdgcn_s_wait_tensorcnt((short)0);
    __syncthreads();                                 // panel kc ready
    if (wave == 0 && kc + 1 < NC)
      tdm_load_64x32(lds_base + (unsigned)(((kc + 1) & 1) * 4096),
                     wbase + (size_t)(kc + 1) * 32); // overlap next DMA
    v16h a = a_next;
    if (kc + 1 < NC) {
      a_next = load_frag_a(arow + (kc + 1) * 32, EMB);
      __builtin_prefetch(arow + (kc + 2) * 32, 0, 1);
    }
    const _Float16* lb = &bt[kc & 1][0];             // 64 rows x 32 k, ld=32
    c0 = wmma16(a, load_frag_b(lb, 32), c0);
    c1 = wmma16(a, load_frag_b(lb + 16 * 32, 32), c1);
    c2 = wmma16(a, load_frag_b(lb + 32 * 32, 32), c2);
    c3 = wmma16(a, load_frag_b(lb + 48 * 32, 32), c3);
    __syncthreads();                                 // all done with panel kc
  }

  const float scaleq = 0.125f;                       // 64^-0.5
  v8f* accs[4] = {&c0, &c1, &c2, &c3};
#pragma unroll
  for (int dt = 0; dt < 4; ++dt) {
    const int e = e0 + dt * 16 + l;                  // output column
    const float bia = bias[e];
    const int h = e >> 6, d = e & 63;
    v8f& c = *accs[dt];
#pragma unroll
    for (int r = 0; r < 8; ++r) {
      const int n = n0 + r + 8 * hiH;                // token row (t*B + b)
      const int t = n >> 1, bb = n & 1;
      const int head = bb * HEADS + h;
      float v = c[r] + bia;
      if (which == 0) {
        qh[((size_t)head * T_SEQ + t) * DHEAD + d] = (_Float16)(v * scaleq);
      } else if (which == 1) {
        kh[((size_t)head * T_SEQ + t) * DHEAD + d] = (_Float16)v;
      } else {
        vt[((size_t)head * DHEAD + d) * T_SEQ + t] = (_Float16)v;
      }
    }
  }
}

// ---------------------------------------------------------------------------
// Flash attention (online softmax).  One wave per (head, 16-query tile).
// Per 32-key step: 4 WMMA scores, fp32 online softmax (shfl_xor + v_exp_f32),
// LDS transpose of P (C-layout -> A-layout, s_wait_dscnt), 4 WMMA for P*V.
// ---------------------------------------------------------------------------
__global__ void flash_attn_kernel(const _Float16* __restrict__ qh,
                                  const _Float16* __restrict__ kh,
                                  const _Float16* __restrict__ vt,
                                  _Float16* __restrict__ attn_h) {
  __shared__ __align__(32) _Float16 lds_p[4][16 * 32];

  const int wave = threadIdx.x >> 5;
  const int gw = blockIdx.x * 4 + wave;
  const int head = gw >> 7;                          // 128 q-tiles per head
  const int qt = gw & 127;
  const int t0 = qt * 16;
  const int lane = threadIdx.x & 31;
  const int l = lane & 15, hiH = lane >> 4;

  const _Float16* qbase = qh + ((size_t)head * T_SEQ + t0) * DHEAD;
  const _Float16* kbase = kh + (size_t)head * T_SEQ * DHEAD;
  const _Float16* vbase = vt + (size_t)head * DHEAD * T_SEQ;

  const v16h a0 = load_frag_a(qbase, DHEAD);         // Q cols 0..31
  const v16h a1 = load_frag_a(qbase + 32, DHEAD);    // Q cols 32..63

  float m[8], lsum[8];
  v8f acc[4] = {v8f{}, v8f{}, v8f{}, v8f{}};
#pragma unroll
  for (int r = 0; r < 8; ++r) { m[r] = -1e30f; lsum[r] = 0.0f; }

  for (int s0 = 0; s0 < T_SEQ; s0 += 32) {
    // ---- scores: S[16 x 32] = Q(16x64) . K^T ----
    v8f sc0 = {}, sc1 = {};
    {
      v16h b = load_frag_b(kbase + (size_t)s0 * DHEAD, DHEAD);
      sc0 = wmma16(a0, b, sc0);
      b = load_frag_b(kbase + (size_t)s0 * DHEAD + 32, DHEAD);
      sc0 = wmma16(a1, b, sc0);
      b = load_frag_b(kbase + (size_t)(s0 + 16) * DHEAD, DHEAD);
      sc1 = wmma16(a0, b, sc1);
      b = load_frag_b(kbase + (size_t)(s0 + 16) * DHEAD + 32, DHEAD);
      sc1 = wmma16(a1, b, sc1);
    }
    if (s0 + 32 < T_SEQ) {
      __builtin_prefetch(kbase + (size_t)(s0 + 32) * DHEAD, 0, 1);
      __builtin_prefetch(vbase + s0 + 32, 0, 1);
    }

    // ---- online softmax (fp32). Row M=r+8*hi spans 16 lanes of one half. ----
#pragma unroll
    for (int r = 0; r < 8; ++r) {
      float mx = fmaxf(sc0[r], sc1[r]);
#pragma unroll
      for (int msk = 1; msk <= 8; msk <<= 1)
        mx = fmaxf(mx, __shfl_xor(mx, msk, 32));
      const float mnew = fmaxf(m[r], mx);
      const float corr = __expf(m[r] - mnew);
      const float p0 = __expf(sc0[r] - mnew);
      const float p1 = __expf(sc1[r] - mnew);
      float rs = p0 + p1;
#pragma unroll
      for (int msk = 1; msk <= 8; msk <<= 1)
        rs += __shfl_xor(rs, msk, 32);
      lsum[r] = lsum[r] * corr + rs;
      m[r] = mnew;
      acc[0][r] *= corr; acc[1][r] *= corr;
      acc[2][r] *= corr; acc[3][r] *= corr;
      // stage P tile (row-major 16x32) in LDS for A-fragment reload
      lds_p[wave][(r + 8 * hiH) * 32 + l]      = (_Float16)p0;
      lds_p[wave][(r + 8 * hiH) * 32 + 16 + l] = (_Float16)p1;
    }
    // intra-wave DS RAW fence + compiler reorder barrier
    asm volatile("s_wait_dscnt 0" ::: "memory");

    const v16h pf = load_frag_a(&lds_p[wave][0], 32);

    // ---- O += P(16x32) . V(32x64) ----
#pragma unroll
    for (int dt = 0; dt < 4; ++dt) {
      v16h bvf = load_frag_b(vbase + (size_t)(dt * 16) * T_SEQ + s0, T_SEQ);
      acc[dt] = wmma16(pf, bvf, acc[dt]);
    }
  }

  // ---- normalize and store to attn_h in [T, B, E] f16 layout ----
  const int bb = head >> 4, h = head & 15;
#pragma unroll
  for (int r = 0; r < 8; ++r) {
    const float inv = 1.0f / lsum[r];
    const int t = t0 + r + 8 * hiH;
    const size_t rowoff = ((size_t)t * BATCH + bb) * EMB + h * DHEAD;
#pragma unroll
    for (int dt = 0; dt < 4; ++dt)
      attn_h[rowoff + dt * 16 + l] = (_Float16)(acc[dt][r] * inv);
  }
}

// ---------------------------------------------------------------------------
// Output projection:  out = attn @ Wo^T + bo  (f32 out), same TDM+LDS scheme.
// ---------------------------------------------------------------------------
__global__ void out_proj_kernel(const _Float16* __restrict__ Ah,
                                const _Float16* __restrict__ Woh,
                                const float* __restrict__ bo,
                                float* __restrict__ out) {
  __shared__ __align__(32) _Float16 bt[2][64 * 32];

  const int wave = threadIdx.x >> 5;
  const int row0 = (blockIdx.x >> 4) * 64;
  const int e0 = (blockIdx.x & 15) * 64;
  const int n0 = row0 + wave * 16;
  const int lane = threadIdx.x & 31;
  const int l = lane & 15, hiH = lane >> 4;
  const unsigned lds_base = (unsigned)(unsigned long long)&bt[0][0];

  const _Float16* arow = Ah + (size_t)n0 * EMB;
  const _Float16* wbase = Woh + (size_t)e0 * EMB;

  v8f c0 = {}, c1 = {}, c2 = {}, c3 = {};

  if (wave == 0) tdm_load_64x32(lds_base, wbase);
  v16h a_next = load_frag_a(arow, EMB);

  const int NC = EMB / 32;
  for (int kc = 0; kc < NC; ++kc) {
    if (wave == 0) __builtin_amdgcn_s_wait_tensorcnt((short)0);
    __syncthreads();
    if (wave == 0 && kc + 1 < NC)
      tdm_load_64x32(lds_base + (unsigned)(((kc + 1) & 1) * 4096),
                     wbase + (size_t)(kc + 1) * 32);
    v16h a = a_next;
    if (kc + 1 < NC) {
      a_next = load_frag_a(arow + (kc + 1) * 32, EMB);
      __builtin_prefetch(arow + (kc + 2) * 32, 0, 1);
    }
    const _Float16* lb = &bt[kc & 1][0];
    c0 = wmma16(a, load_frag_b(lb, 32), c0);
    c1 = wmma16(a, load_frag_b(lb + 16 * 32, 32), c1);
    c2 = wmma16(a, load_frag_b(lb + 32 * 32, 32), c2);
    c3 = wmma16(a, load_frag_b(lb + 48 * 32, 32), c3);
    __syncthreads();
  }

  v8f* accs[4] = {&c0, &c1, &c2, &c3};
#pragma unroll
  for (int dt = 0; dt < 4; ++dt) {
    const int e = e0 + dt * 16 + l;
    const float bia = bo[e];
    v8f& c = *accs[dt];
#pragma unroll
    for (int r = 0; r < 8; ++r) {
      const int n = n0 + r + 8 * hiH;
      out[(size_t)n * EMB + e] = c[r] + bia;
    }
  }
}

// ---------------------------------------------------------------------------
// Host-side launcher
// ---------------------------------------------------------------------------
extern "C" void kernel_launch(void* const* d_in, const int* in_sizes, int n_in,
                              void* d_out, int out_size, void* d_ws, size_t ws_size,
                              hipStream_t stream) {
  (void)in_sizes; (void)n_in; (void)out_size; (void)ws_size;

  const float* X  = (const float*)d_in[0];   // [T,B,E]
  const float* Wq = (const float*)d_in[1];
  const float* bq = (const float*)d_in[2];
  const float* Wk = (const float*)d_in[3];
  const float* bk = (const float*)d_in[4];
  const float* Wv = (const float*)d_in[5];
  const float* bv = (const float*)d_in[6];
  const float* Wo = (const float*)d_in[7];
  const float* bo = (const float*)d_in[8];
  float* out = (float*)d_out;

  const int NE = NROWS * EMB;     // 4M token elements
  const int WE = EMB * EMB;       // 1M weight elements

  _Float16* ws  = (_Float16*)d_ws;
  _Float16* Xh  = ws;             // 4M
  _Float16* Wqh = Xh + NE;        // 1M
  _Float16* Wkh = Wqh + WE;
  _Float16* Wvh = Wkh + WE;
  _Float16* Woh = Wvh + WE;
  _Float16* qh  = Woh + WE;       // [32][2048][64]
  _Float16* kh  = qh + NE;        // [32][2048][64]
  _Float16* vt  = kh + NE;        // [32][64][2048]
  _Float16* ah  = vt + NE;        // [T,B,E] f16

  // stage 0: fp32 -> fp16 conversion (all on-chip afterwards; fits in L2)
  f32_to_f16_kernel<<<2048, 256, 0, stream>>>(X,  Xh,  NE);
  f32_to_f16_kernel<<<1024, 256, 0, stream>>>(Wq, Wqh, WE);
  f32_to_f16_kernel<<<1024, 256, 0, stream>>>(Wk, Wkh, WE);
  f32_to_f16_kernel<<<1024, 256, 0, stream>>>(Wv, Wvh, WE);
  f32_to_f16_kernel<<<1024, 256, 0, stream>>>(Wo, Woh, WE);

  // stage 1: fused QKV projection (grid.y selects Q/K/V)
  qkv_gemm_kernel<<<dim3(1024, 3), 128, 0, stream>>>(
      Xh, Wqh, Wkh, Wvh, bq, bk, bv, qh, kh, vt);

  // stage 2: flash attention
  flash_attn_kernel<<<1024, 128, 0, stream>>>(qh, kh, vt, ah);

  // stage 3: output projection
  out_proj_kernel<<<1024, 128, 0, stream>>>(ah, Woh, bo, out);
}
